// LlamaBlock_10917806866861
// MI455X (gfx1250) — compile-verified, matrix operand order checked by reading, not run on hardware
//
#include <hip/hip_runtime.h>

// ---------------------------------------------------------------------------
// LlamaBlock for MI455X (gfx1250, wave32, WMMA).
// B=8, M=1024, D=512, H=8 (head dim = 512), BM = 8192, HD = 4096.
// All GEMMs on v_wmma_f32_16x16x32_bf16 (f32 accumulate); flash attention
// (no 256MB score tensor); async global->LDS staging in the attention kernel.
// ---------------------------------------------------------------------------

#define CB   8
#define CM   1024
#define CD   512
#define CH   8
#define CBM  (CB * CM)
#define CHD  (CH * CD)

#ifndef USE_ASYNC_LDS
#define USE_ASYNC_LDS 1
#endif

typedef __attribute__((ext_vector_type(16))) __bf16 v16bf;
typedef __attribute__((ext_vector_type(8)))  float  v8f;

__device__ __forceinline__ v8f wmma_bf16(v16bf a, v16bf b, v8f c) {
  return __builtin_amdgcn_wmma_f32_16x16x32_bf16(
      /*neg_a=*/false, a, /*neg_b=*/false, b,
      /*c_mod=*/(short)0, c, /*reuse_a=*/false, /*reuse_b=*/false);
}

// 16x32 bf16 A-layout fragment (also B operand of an NT product = rows of
// B^T). Per ISA 7.12.2 the per-lane data is two contiguous 16B groups:
//   half = lane>>4 : VGPR0-3 = K[8h .. 8h+7], VGPR4-7 = K[16+8h .. 23+8h]
// -> exactly two ds_load_b128 per lane. Requires ldt % 8 == 0 (16B rows).
__device__ __forceinline__ v16bf load_frag(const __bf16* base, int ldt) {
  int lane = threadIdx.x & 31;
  const __bf16* p = base + (lane & 15) * ldt + 8 * (lane >> 4);
  union { v16bf v; uint4 q[2]; } f;
  f.q[0] = *(const uint4*)(p);       // K group 0
  f.q[1] = *(const uint4*)(p + 16);  // K group 1 (+32B)
  return f.v;
}

__device__ __forceinline__ v8f vzero8() {
  v8f z;
#pragma unroll
  for (int i = 0; i < 8; ++i) z[i] = 0.0f;
  return z;
}

// Async global -> LDS copy (ASYNCcnt-tracked, no VGPR data round trip).
__device__ __forceinline__ void async_copy_b128(__bf16* lds_dst, const __bf16* gsrc) {
#if USE_ASYNC_LDS
  unsigned ldsoff = (unsigned)(unsigned long long)lds_dst;  // addr[31:0] = LDS offset
  unsigned long long ga = (unsigned long long)gsrc;
  asm volatile("global_load_async_to_lds_b128 %0, %1, off"
               :: "v"(ldsoff), "v"(ga) : "memory");
#else
  *(uint4*)lds_dst = *(const uint4*)gsrc;
#endif
}

__device__ __forceinline__ void async_join() {
#if USE_ASYNC_LDS
  asm volatile("s_wait_asynccnt 0" ::: "memory");
#endif
}

// ---------------------------------------------------------------------------
// Elementwise helpers
// ---------------------------------------------------------------------------
__global__ void k_to_bf16(const float* __restrict__ in, __bf16* __restrict__ out, int n) {
  int i = blockIdx.x * blockDim.x + threadIdx.x;
  if (i < n) out[i] = (__bf16)in[i];
}

__global__ __launch_bounds__(256) void k_rms_reduce(const float* __restrict__ x,
                                                    float* __restrict__ rms, int perB) {
  __shared__ float red[256];
  const float4* xb = (const float4*)(x + (long long)blockIdx.x * perB);
  float s = 0.0f;
  for (int i = threadIdx.x; i < perB / 4; i += 256) {
    float4 v = xb[i];
    s += v.x * v.x + v.y * v.y + v.z * v.z + v.w * v.w;
  }
  red[threadIdx.x] = s;
  __syncthreads();
  for (int st = 128; st > 0; st >>= 1) {
    if (threadIdx.x < st) red[threadIdx.x] += red[threadIdx.x + st];
    __syncthreads();
  }
  if (threadIdx.x == 0) rms[blockIdx.x] = sqrtf(red[0]) * rsqrtf((float)perB);
}

__global__ void k_rms_apply(const float* __restrict__ x, const float* __restrict__ scale,
                            const float* __restrict__ rms, float* __restrict__ xn,
                            __bf16* __restrict__ xnb, int perB) {
  int i = blockIdx.x * blockDim.x + threadIdx.x;
  int b = i / perB, md = i - b * perB;
  float v = scale[md] * x[i] * (1.0f / rms[b]);
  xn[i] = v;
  xnb[i] = (__bf16)v;
}

__global__ void k_swiglu_out(const float* __restrict__ x3, const float* __restrict__ g,
                             const float* __restrict__ l, const float* __restrict__ beta,
                             float* __restrict__ out, int n) {
  int i = blockIdx.x * blockDim.x + threadIdx.x;
  if (i >= n) return;
  float gv = g[i];
  float sw = gv / (1.0f + __expf(-beta[0] * gv));
  out[i] = x3[i] + sw * l[i];
}

// ---------------------------------------------------------------------------
// Generic NT GEMM: C[M x N] = A[M x K] * W[N x K]^T  (+ epilogue)
// 128x128 tile / 256-thread block (8 waves; wave = 64x32). Software-pipelined:
// tile k+1 is loaded into VGPRs while WMMAs consume tile k from LDS. All six
// fragments of a k-step are loaded before the 8 WMMAs so the scheduler can
// batch the DS waits and issue WMMAs back-to-back.
// ---------------------------------------------------------------------------
enum { EPI_F32 = 0, EPI_BF16_BIAS = 1, EPI_ROPE = 2, EPI_TRANS = 3 };

template <int EPI>
__global__ __launch_bounds__(256) void k_gemm_nt(
    const __bf16* __restrict__ A, int lda, long long strideAz,
    const __bf16* __restrict__ W, int ldw, long long strideWz,
    int K,
    const float* __restrict__ bias, const float* __restrict__ resid, int ldr,
    void* __restrict__ outv, int ldo, long long strideOz,
    int Bz, int ropeD) {
  __shared__ __bf16 As[128 * 40];  // 80B rows -> b128-aligned
  __shared__ __bf16 Ws[128 * 40];

  const int z = blockIdx.z;
  const int bidx = z % Bz, hidx = z / Bz;
  A += (long long)bidx * strideAz;
  W += (long long)hidx * strideWz;

  const int m0 = blockIdx.x * 128;
  const int n0 = blockIdx.y * 128;
  const int t = threadIdx.x;
  const int wave = t >> 5;
  const int wm = wave >> 2;  // 0..1 : 64 rows
  const int wn = wave & 3;   // 0..3 : 32 cols

  v8f acc[4][2];
#pragma unroll
  for (int mt = 0; mt < 4; ++mt)
#pragma unroll
    for (int nt = 0; nt < 2; ++nt) acc[mt][nt] = vzero8();

  uint4 pa[2], pw[2];
  auto load_tiles = [&](int k0) {
#pragma unroll
    for (int r = 0; r < 2; ++r) {
      int idx = t + 256 * r;  // 0..511
      int row = idx >> 2;     // 0..127
      int ch  = idx & 3;      // 4 chunks of 8 bf16
      pa[r] = *(const uint4*)(A + (long long)(m0 + row) * lda + k0 + ch * 8);
      pw[r] = *(const uint4*)(W + (long long)(n0 + row) * ldw + k0 + ch * 8);
    }
  };
  load_tiles(0);

  for (int k0 = 0; k0 < K; k0 += 32) {
    __syncthreads();  // previous iteration's WMMAs done with LDS
#pragma unroll
    for (int r = 0; r < 2; ++r) {
      int idx = t + 256 * r;
      int row = idx >> 2, ch = idx & 3;
      *(uint4*)(As + row * 40 + ch * 8) = pa[r];
      *(uint4*)(Ws + row * 40 + ch * 8) = pw[r];
    }
    __syncthreads();
    if (k0 + 32 < K) load_tiles(k0 + 32);  // overlaps with WMMAs below
    if (k0 + 64 < K) {                      // L2/WGP$ prefetch 2 tiles ahead
      int row = t >> 2, ch = t & 3;
      __builtin_prefetch(A + (long long)(m0 + row) * lda + k0 + 64 + ch * 8, 0, 1);
      __builtin_prefetch(W + (long long)(n0 + row) * ldw + k0 + 64 + ch * 8, 0, 1);
    }

    // batch all fragment loads, then run the 8 WMMAs back-to-back
    v16bf bfr[2], afr[4];
#pragma unroll
    for (int nt = 0; nt < 2; ++nt)
      bfr[nt] = load_frag(Ws + (wn * 32 + nt * 16) * 40, 40);
#pragma unroll
    for (int mt = 0; mt < 4; ++mt)
      afr[mt] = load_frag(As + (wm * 64 + mt * 16) * 40, 40);
#pragma unroll
    for (int mt = 0; mt < 4; ++mt)
#pragma unroll
      for (int nt = 0; nt < 2; ++nt)
        acc[mt][nt] = wmma_bf16(afr[mt], bfr[nt], acc[mt][nt]);
  }

  // epilogue: C-frag: VGPR j -> m = j + 8*(lane>>4), n = lane&15
  const int lane = t & 31, half = lane >> 4, lx = lane & 15;
#pragma unroll
  for (int mt = 0; mt < 4; ++mt) {
#pragma unroll
    for (int nt = 0; nt < 2; ++nt) {
      const int nn = n0 + wn * 32 + nt * 16 + lx;
#pragma unroll
      for (int j = 0; j < 8; ++j) {
        const int mm = m0 + wm * 64 + mt * 16 + j + 8 * half;
        float v = acc[mt][nt][j];
        if (EPI == EPI_F32) {
          if (bias)  v += bias[nn];
          if (resid) v += resid[(long long)mm * ldr + nn];
          ((float*)outv)[(long long)z * strideOz + (long long)mm * ldo + nn] = v;
        } else if (EPI == EPI_BF16_BIAS) {
          v += bias[nn];
          ((__bf16*)outv)[(long long)z * strideOz + (long long)mm * ldo + nn] = (__bf16)v;
        } else if (EPI == EPI_ROPE) {
          float pv = __shfl_xor(v, 1, 32);  // adjacent head-dims in adjacent lanes
          float i2 = (float)(nn >> 1);
          float theta = __powf(10000.0f, -2.0f * (i2 - 1.0f) / (float)ropeD);
          float s, c;
          __sincosf((float)mm * theta, &s, &c);
          float te = (nn & 1) ? pv : v;
          float to = (nn & 1) ? v : pv;
          float r = (nn & 1) ? (-te * s + to * c) : (te * c + to * s);
          ((__bf16*)outv)[(long long)z * strideOz + (long long)mm * ldo + nn] = (__bf16)r;
        } else {  // EPI_TRANS: out[n][m] (producing V^T)
          ((__bf16*)outv)[(long long)z * strideOz + (long long)nn * ldo + mm] = (__bf16)v;
        }
      }
    }
  }
}

// ---------------------------------------------------------------------------
// Flash attention: per (h,b, 64-row q tile). 512 threads = 16 waves.
// S phase: wave (wq,wk) computes one 16x16 S frag (K=512).
// O phase: wave w owns all 64 q rows x 32 d-cols -> 8 C frags in registers.
// K/V/Q staging uses async global->LDS (ASYNCcnt).
// ---------------------------------------------------------------------------
#define LDQ 520   // 1040B rows (16B aligned, conflict-free frag loads)
#define LDV 72    // 144B rows
#define LDSS 68

__global__ __launch_bounds__(512) void k_flash_attn(
    const __bf16* __restrict__ Q, const __bf16* __restrict__ Kr,
    const __bf16* __restrict__ Vt, __bf16* __restrict__ ctx, int Bz) {
  extern __shared__ char smem[];
  __bf16* Qs = (__bf16*)smem;            // 64 x LDQ
  __bf16* Ks = Qs + 64 * LDQ;            // 64 x LDQ
  __bf16* Vs = Ks + 64 * LDQ;            // 512 x LDV  (rows = d, cols = k)
  __bf16* Ps = Vs + 512 * LDV;           // 64 x LDV
  float*  Ss = (float*)(Ps + 64 * LDV);  // 64 x LDSS
  float* rowmax   = Ss + 64 * LDSS;
  float* rowsum   = rowmax + 64;
  float* rowscale = rowsum + 64;

  const int z = blockIdx.y;
  const int b = z % Bz, h = z / Bz;
  Q  += (long long)z * CM * CD;
  Kr += (long long)z * CM * CD;
  Vt += (long long)z * CD * CM;
  const int q0 = blockIdx.x * 64;
  const int t = threadIdx.x;
  const int wave = t >> 5, lane = t & 31, half = lane >> 4, lx = lane & 15;
  const int wq = wave >> 2, wk = wave & 3;  // S-phase 4x4 wave grid
  const int dw = wave * 32;                 // O-phase d-col base

  // Q tile resident in LDS (64 x 512 bf16), async DMA
#pragma unroll
  for (int r = 0; r < 8; ++r) {
    int idx = t + 512 * r;
    int row = idx >> 6, ch = idx & 63;
    async_copy_b128(Qs + row * LDQ + ch * 8,
                    Q + (long long)(q0 + row) * CD + ch * 8);
  }
  if (t < 64) { rowmax[t] = -1e30f; rowsum[t] = 0.0f; }

  v8f o[4][2];
#pragma unroll
  for (int mt = 0; mt < 4; ++mt) { o[mt][0] = vzero8(); o[mt][1] = vzero8(); }
  async_join();
  __syncthreads();

  for (int kt = 0; kt < CM / 64; ++kt) {
    const int k0r = kt * 64;
    // stage K tile (64x512) and V^T tile (512x64) via async DMA
#pragma unroll
    for (int r = 0; r < 8; ++r) {
      int idx = t + 512 * r;
      int row = idx >> 6, ch = idx & 63;
      async_copy_b128(Ks + row * LDQ + ch * 8,
                      Kr + (long long)(k0r + row) * CD + ch * 8);
      int vr = idx >> 3, vc = idx & 7;
      async_copy_b128(Vs + vr * LDV + vc * 8,
                      Vt + (long long)vr * CM + k0r + vc * 8);
    }
    async_join();
    __syncthreads();

    // S = (Q K^T) * 1/sqrt(D)
    {
      v8f s = vzero8();
#pragma unroll
      for (int kk = 0; kk < CD; kk += 32) {
        v16bf a  = load_frag(Qs + (wq * 16) * LDQ + kk, LDQ);
        v16bf bb = load_frag(Ks + (wk * 16) * LDQ + kk, LDQ);
        s = wmma_bf16(a, bb, s);
      }
#pragma unroll
      for (int j = 0; j < 8; ++j) {
        int mm = wq * 16 + j + 8 * half;
        int nn = wk * 16 + lx;
        Ss[mm * LDSS + nn] = s[j] * 0.04419417382415922f;  // 1/sqrt(512)
      }
    }
    __syncthreads();

    // online softmax: 8 lanes per row
    {
      int row = t >> 3, g = t & 7;
      float vals[8];
      float mx = -1e30f;
#pragma unroll
      for (int i = 0; i < 8; ++i) {
        vals[i] = Ss[row * LDSS + g * 8 + i];
        mx = fmaxf(mx, vals[i]);
      }
#pragma unroll
      for (int d2 = 1; d2 < 8; d2 <<= 1) mx = fmaxf(mx, __shfl_xor(mx, d2, 32));
      float mold = rowmax[row];
      float mnew = fmaxf(mold, mx);
      float ps = 0.0f;
#pragma unroll
      for (int i = 0; i < 8; ++i) {
        float p = __expf(vals[i] - mnew);
        Ps[row * LDV + g * 8 + i] = (__bf16)p;
        ps += p;
      }
#pragma unroll
      for (int d2 = 1; d2 < 8; d2 <<= 1) ps += __shfl_xor(ps, d2, 32);
      if (g == 0) {
        float corr = __expf(mold - mnew);  // 0 on first tile
        rowscale[row] = corr;
        rowsum[row] = rowsum[row] * corr + ps;
        rowmax[row] = mnew;
      }
    }
    __syncthreads();

    // rescale O, then O += P * V (fragments batched before WMMAs)
#pragma unroll
    for (int mt = 0; mt < 4; ++mt)
#pragma unroll
      for (int j = 0; j < 8; ++j) {
        float sc = rowscale[mt * 16 + j + 8 * half];
        o[mt][0][j] *= sc;
        o[mt][1][j] *= sc;
      }
#pragma unroll
    for (int kk = 0; kk < 64; kk += 32) {
      v16bf bv0 = load_frag(Vs + (dw + 0)  * LDV + kk, LDV);
      v16bf bv1 = load_frag(Vs + (dw + 16) * LDV + kk, LDV);
      v16bf ap[4];
#pragma unroll
      for (int mt = 0; mt < 4; ++mt)
        ap[mt] = load_frag(Ps + (mt * 16) * LDV + kk, LDV);
#pragma unroll
      for (int mt = 0; mt < 4; ++mt) {
        o[mt][0] = wmma_bf16(ap[mt], bv0, o[mt][0]);
        o[mt][1] = wmma_bf16(ap[mt], bv1, o[mt][1]);
      }
    }
    __syncthreads();
  }

  // write ctx into cat layout: cat[b, m, h*D + d]
#pragma unroll
  for (int mt = 0; mt < 4; ++mt) {
#pragma unroll
    for (int j = 0; j < 8; ++j) {
      int mm = mt * 16 + j + 8 * half;
      float inv = 1.0f / rowsum[mm];
#pragma unroll
      for (int nt = 0; nt < 2; ++nt) {
        int dd = dw + nt * 16 + lx;
        ctx[(long long)(b * CM + q0 + mm) * CHD + h * CD + dd] =
            (__bf16)(o[mt][nt][j] * inv);
      }
    }
  }
}

// ---------------------------------------------------------------------------
// Host-side orchestration
// ---------------------------------------------------------------------------
extern "C" void kernel_launch(void* const* d_in, const int* in_sizes, int n_in,
                              void* d_out, int out_size, void* d_ws, size_t ws_size,
                              hipStream_t stream) {
  (void)in_sizes; (void)n_in; (void)out_size; (void)ws_size;
  const float* x     = (const float*)d_in[0];
  const float* scale = (const float*)d_in[1];
  const float* Wq    = (const float*)d_in[2];
  const float* Wk    = (const float*)d_in[3];
  const float* Wv    = (const float*)d_in[4];
  const float* Wo    = (const float*)d_in[5];
  const float* bo    = (const float*)d_in[6];
  const float* Wff   = (const float*)d_in[7];
  const float* bff   = (const float*)d_in[8];
  const float* Wg    = (const float*)d_in[9];
  const float* bg    = (const float*)d_in[10];
  const float* Wl    = (const float*)d_in[11];
  const float* bl    = (const float*)d_in[12];
  const float* beta  = (const float*)d_in[13];
  float* out = (float*)d_out;

  char* w = (char*)d_ws;
  size_t off = 0;
  auto alloc = [&](size_t bytes) -> void* {
    void* p = w + off;
    off = (off + bytes + 255) & ~(size_t)255;
    return p;
  };
  const size_t BMD  = (size_t)CBM * CD;
  const size_t HBMD = (size_t)CH * CB * CM * CD;

  float*  rmsA  = (float*)alloc(CB * sizeof(float));
  float*  rmsB  = (float*)alloc(CB * sizeof(float));
  float*  xn_f  = (float*)alloc(BMD * 4);
  float*  x2_f  = (float*)alloc(BMD * 4);
  float*  x3_f  = (float*)alloc(BMD * 4);
  float*  g_f   = (float*)alloc(BMD * 4);
  float*  l_f   = (float*)alloc(BMD * 4);
  __bf16* xn_b  = (__bf16*)alloc(BMD * 2);
  __bf16* x3_b  = (__bf16*)alloc(BMD * 2);
  __bf16* h1_b  = (__bf16*)alloc(BMD * 2);
  __bf16* qr_b  = (__bf16*)alloc(HBMD * 2);
  __bf16* kr_b  = (__bf16*)alloc(HBMD * 2);
  __bf16* vT_b  = (__bf16*)alloc(HBMD * 2);
  __bf16* ctx_b = (__bf16*)alloc((size_t)CBM * CHD * 2);
  __bf16* Wq_b  = (__bf16*)alloc((size_t)CH * CD * CD * 2);
  __bf16* Wk_b  = (__bf16*)alloc((size_t)CH * CD * CD * 2);
  __bf16* Wv_b  = (__bf16*)alloc((size_t)CH * CD * CD * 2);
  __bf16* Wo_b  = (__bf16*)alloc((size_t)CD * CHD * 2);
  __bf16* Wff_b = (__bf16*)alloc((size_t)CD * CD * 2);
  __bf16* Wg_b  = (__bf16*)alloc((size_t)CD * CD * 2);
  __bf16* Wl_b  = (__bf16*)alloc((size_t)CD * CD * 2);

  const int perB = CM * CD;
  auto cvt = [&](const float* src, __bf16* dst, int n) {
    k_to_bf16<<<(n + 255) / 256, 256, 0, stream>>>(src, dst, n);
  };
  cvt(Wq,  Wq_b,  CH * CD * CD);
  cvt(Wk,  Wk_b,  CH * CD * CD);
  cvt(Wv,  Wv_b,  CH * CD * CD);
  cvt(Wo,  Wo_b,  CD * CHD);
  cvt(Wff, Wff_b, CD * CD);
  cvt(Wg,  Wg_b,  CD * CD);
  cvt(Wl,  Wl_b,  CD * CD);

  // xn = rmsnorm(x)
  k_rms_reduce<<<CB, 256, 0, stream>>>(x, rmsA, perB);
  k_rms_apply<<<(int)(BMD / 256), 256, 0, stream>>>(x, scale, rmsA, xn_f, xn_b, perB);

  // QKV (+RoPE, V transposed), batched over z = h*B + b
  dim3 gQKV(CM / 128, CD / 128, CH * CB);
  k_gemm_nt<EPI_ROPE><<<gQKV, 256, 0, stream>>>(
      xn_b, CD, (long long)CM * CD, Wq_b, CD, (long long)CD * CD, CD,
      nullptr, nullptr, 0, qr_b, CD, (long long)CM * CD, CB, CD);
  k_gemm_nt<EPI_ROPE><<<gQKV, 256, 0, stream>>>(
      xn_b, CD, (long long)CM * CD, Wk_b, CD, (long long)CD * CD, CD,
      nullptr, nullptr, 0, kr_b, CD, (long long)CM * CD, CB, CD);
  k_gemm_nt<EPI_TRANS><<<gQKV, 256, 0, stream>>>(
      xn_b, CD, (long long)CM * CD, Wv_b, CD, (long long)CD * CD, CD,
      nullptr, nullptr, 0, vT_b, CM, (long long)CD * CM, CB, CD);

  // flash attention -> ctx in cat layout
  size_t smem = (size_t)(2 * 64 * LDQ + 512 * LDV + 64 * LDV) * 2 +
                (size_t)(64 * LDSS + 192) * 4;
  k_flash_attn<<<dim3(CM / 64, CH * CB), 512, smem, stream>>>(qr_b, kr_b, vT_b, ctx_b, CB);

  // x2 = xn + cat @ Wo^T + bo  (K = 4096)
  k_gemm_nt<EPI_F32><<<dim3(CBM / 128, CD / 128, 1), 256, 0, stream>>>(
      ctx_b, CHD, 0, Wo_b, CHD, 0, CHD,
      bo, xn_f, CD, x2_f, CD, 0, 1, 0);

  // x3 = rmsnorm(x2)
  k_rms_reduce<<<CB, 256, 0, stream>>>(x2_f, rmsB, perB);
  k_rms_apply<<<(int)(BMD / 256), 256, 0, stream>>>(x2_f, scale, rmsB, x3_f, x3_b, perB);

  // h1 = x3 @ Wff^T + bff (bf16)
  k_gemm_nt<EPI_BF16_BIAS><<<dim3(CBM / 128, CD / 128, 1), 256, 0, stream>>>(
      x3_b, CD, 0, Wff_b, CD, 0, CD,
      bff, nullptr, 0, h1_b, CD, 0, 1, 0);

  // g = h1 @ Wg^T + bg ; l = h1 @ Wl^T + bl
  k_gemm_nt<EPI_F32><<<dim3(CBM / 128, CD / 128, 1), 256, 0, stream>>>(
      h1_b, CD, 0, Wg_b, CD, 0, CD, bg, nullptr, 0, g_f, CD, 0, 1, 0);
  k_gemm_nt<EPI_F32><<<dim3(CBM / 128, CD / 128, 1), 256, 0, stream>>>(
      h1_b, CD, 0, Wl_b, CD, 0, CD, bl, nullptr, 0, l_f, CD, 0, 1, 0);

  // out = x3 + swish(g) * l
  k_swiglu_out<<<(int)(BMD / 256), 256, 0, stream>>>(x3_f, g_f, l_f, beta, out, (int)BMD);
}
